// ANI2x_74079595921544
// MI455X (gfx1250) — compile-verified, matrix operand order checked by reading, not run on hardware
//
#include <hip/hip_runtime.h>
#include <cstdint>
#include <cstddef>

// ---------------------------------------------------------------------------
// ANI-2x energy + forces for MI455X (gfx1250, wave32, WMMA).
// MLP fwd+bwd runs on v_wmma_f32_16x16x32_f16 (f16 in, f32 accum), atoms
// grouped by species into 16-row tiles. B-operand weights are pre-packed in
// WMMA fragment order so each K-step is two b128 loads per operand.
// AEV fwd/bwd are VALU kernels with LDS-resident neighbor data.
// ---------------------------------------------------------------------------

typedef __attribute__((ext_vector_type(16))) _Float16 v16h;
typedef __attribute__((ext_vector_type(8)))  _Float16 v8h;
typedef __attribute__((ext_vector_type(8)))  float    v8f;

#define PI_F 3.14159265358979323846f

#define NSPC     7
#define AEV_PAD  1024      // 1008 real cols, padded to multiple of 32 for K
#define RAD_OFF  112       // 7 species * 16 radial shifts
#define ANG_LEN  896       // 28 pair bins * 32 (4 zeta * 8 radial) shifts
#define H1D      256
#define H2D      192
#define H3D      160
#define KNBR     32
#define NPAIR    496       // 32 choose 2
#define MAX_TILES 144      // ceil(2048/16) + 7 species of padding <= 135

// model constants
#define RCR_C  5.1f
#define RCA_C  3.5f
#define ETAR_C 19.7f
#define ETAA_C 12.5f
#define ZETA_C 14.1f

__device__ __forceinline__ int clamp_sp(int s) { return s < 0 ? 0 : (s > 6 ? 6 : s); }

// ---------------------------------------------------------------------------
// Pack weights (fp32 -> fp16) directly into WMMA B-fragment order.
// Output linear layout: ((((s*nNT + nt)*nK0 + ks)*32 + lane)*16 + e)
// where for lane/e the source element is
//   v = e>>1 ; k = ks*32 + ((v&4)?16:0) + (lane>>4)*8 + 2*(v&3) + (e&1)
//   n = nt*16 + (lane&15)
// trans==0: B[k][n] = W[s][k][n] (W is [S][R=Kreal][C=Ncols])
// trans==1: B[k][n] = W[s][n][k] (W is [S][R=Ncols][C=Kreal])
// k >= Kreal pads with zero.
// ---------------------------------------------------------------------------
__global__ void k_packB(const float* __restrict__ W, _Float16* __restrict__ out,
                        int S, int R, int C, int trans, int Kreal, int nK0, int nNT) {
    size_t idx = (size_t)blockIdx.x * blockDim.x + threadIdx.x;
    size_t total = (size_t)S * nNT * nK0 * 32 * 16;
    if (idx >= total) return;
    int e = (int)(idx & 15);
    int lane = (int)((idx >> 4) & 31);
    size_t t = idx >> 9;
    int ks = (int)(t % nK0); t /= nK0;
    int nt = (int)(t % nNT);
    int s = (int)(t / nNT);
    int v = e >> 1;
    int k = ks * 32 + ((v & 4) ? 16 : 0) + (lane >> 4) * 8 + 2 * (v & 3) + (e & 1);
    int n = nt * 16 + (lane & 15);
    float val = 0.0f;
    if (k < Kreal) {
        val = trans ? W[((size_t)s * R + n) * C + k]
                    : W[((size_t)s * R + k) * C + n];
    }
    out[idx] = (_Float16)val;
}

__global__ void k_f2h(const float* __restrict__ in, _Float16* __restrict__ out, size_t total) {
    size_t idx = (size_t)blockIdx.x * blockDim.x + threadIdx.x;
    if (idx >= total) return;
    out[idx] = (_Float16)in[idx];
}

// ---------------------------------------------------------------------------
// radial AEV forward: one thread per edge, scatter into aevF via atomics
// ---------------------------------------------------------------------------
__global__ void k_radial_fwd(const float* __restrict__ c, const int* __restrict__ sp,
                             const int* __restrict__ ai, const unsigned char* __restrict__ em,
                             float* __restrict__ aevF, int E) {
    int e = blockIdx.x * blockDim.x + threadIdx.x;
    if (e >= E || !em[e]) return;
    int i = ai[e], j = ai[E + e];
    float dx = c[3 * j + 0] - c[3 * i + 0];
    float dy = c[3 * j + 1] - c[3 * i + 1];
    float dz = c[3 * j + 2] - c[3 * i + 2];
    float d = sqrtf(dx * dx + dy * dy + dz * dz + 1e-12f);
    if (d >= RCR_C) return;
    float fc = 0.5f * cosf(PI_F * d / RCR_C) + 0.5f;
    int si = sp[i], sj = sp[j];
    float* ri = aevF + (size_t)i * AEV_PAD + sj * 16;
    float* rj = aevF + (size_t)j * AEV_PAD + si * 16;
#pragma unroll
    for (int m = 0; m < 16; ++m) {
        float t = d - (0.8f + 0.26875f * m);
        float r = 0.25f * expf(-ETAR_C * t * t) * fc;
        atomicAdd(ri + m, r);
        atomicAdd(rj + m, r);
    }
}

// ---------------------------------------------------------------------------
// angular AEV forward: one block per atom, LDS-resident neighbors + bins
// ---------------------------------------------------------------------------
__global__ void k_angular_fwd(const float* __restrict__ c, const int* __restrict__ sp,
                              const int* __restrict__ nbrj, const unsigned char* __restrict__ nm,
                              float* __restrict__ aevF) {
    int n = blockIdx.x;
    int tid = threadIdx.x;
    __shared__ float vx[KNBR], vy[KNBR], vz[KNBR], dn[KNBR], fc[KNBR];
    __shared__ int spj[KNBR];
    __shared__ float acc[ANG_LEN];

    if (tid < KNBR) {
        int j = nbrj[n * KNBR + tid];
        float x = c[3 * j + 0] - c[3 * n + 0];
        float y = c[3 * j + 1] - c[3 * n + 1];
        float z = c[3 * j + 2] - c[3 * n + 2];
        float d = sqrtf(x * x + y * y + z * z + 1e-12f);
        float f = (d < RCA_C) ? (0.5f * cosf(PI_F * d / RCA_C) + 0.5f) : 0.0f;
        if (!nm[n * KNBR + tid]) f = 0.0f;
        vx[tid] = x; vy[tid] = y; vz[tid] = z; dn[tid] = d; fc[tid] = f;
        spj[tid] = sp[j];
    }
    for (int idx = tid; idx < ANG_LEN; idx += blockDim.x) acc[idx] = 0.0f;
    __syncthreads();

    for (int pid = tid; pid < NPAIR; pid += blockDim.x) {
        int rem = pid, jj = 0;
        while (rem >= (KNBR - 1) - jj) { rem -= (KNBR - 1) - jj; ++jj; }
        int kk = jj + 1 + rem;
        float bse = 2.0f * fc[jj] * fc[kk];
        if (bse == 0.0f) continue;
        float dj = dn[jj], dk = dn[kk];
        float dot = vx[jj] * vx[kk] + vy[jj] * vy[kk] + vz[jj] * vz[kk];
        float u = dot / fmaxf(dj * dk, 1e-10f);
        float th = acosf(0.95f * u);
        float davg = 0.5f * (dj + dk);
        int a = min(spj[jj], spj[kk]);
        int b = max(spj[jj], spj[kk]);
        int p = a * NSPC - (a * (a - 1)) / 2 + (b - a);
        float f2[8];
#pragma unroll
        for (int aa = 0; aa < 8; ++aa) {
            float t = davg - (0.8f + 0.3375f * aa);
            f2[aa] = expf(-ETAA_C * t * t);
        }
#pragma unroll
        for (int z = 0; z < 4; ++z) {
            float hz = 0.5f * (1.0f + cosf(th - (z + 0.5f) * (PI_F / 4.0f)));
            float f1 = powf(fmaxf(hz, 0.0f), ZETA_C);
            float bf1 = bse * f1;
#pragma unroll
            for (int aa = 0; aa < 8; ++aa)
                atomicAdd(&acc[p * 32 + z * 8 + aa], bf1 * f2[aa]);
        }
    }
    __syncthreads();
    for (int idx = tid; idx < ANG_LEN; idx += blockDim.x)
        aevF[(size_t)n * AEV_PAD + RAD_OFF + idx] = acc[idx];
}

// ---------------------------------------------------------------------------
// species bucketing: counts -> offsets + tile table -> atom lists
// ---------------------------------------------------------------------------
__global__ void k_count(const int* __restrict__ sp, int* counts, int N) {
    int a = blockIdx.x * blockDim.x + threadIdx.x;
    if (a >= N) return;
    atomicAdd(&counts[clamp_sp(sp[a])], 1);
}

__global__ void k_build_tiles(const int* __restrict__ counts, int* offsets,
                              int* tileS, int* tileBase, int* nTiles) {
    if (threadIdx.x != 0 || blockIdx.x != 0) return;
    int off = 0, nt = 0;
    for (int s = 0; s < NSPC; ++s) {
        offsets[s] = off;
        int cnt = counts[s];
        int t = (cnt + 15) / 16;
        for (int i = 0; i < t && nt < MAX_TILES; ++i) {
            tileS[nt] = s;
            tileBase[nt] = off + 16 * i;
            ++nt;
        }
        off += cnt;
    }
    nTiles[0] = nt;
}

__global__ void k_fill(const int* __restrict__ sp, const int* __restrict__ offsets,
                       int* cursor, int* atomList, int N) {
    int a = blockIdx.x * blockDim.x + threadIdx.x;
    if (a >= N) return;
    int s = clamp_sp(sp[a]);
    int p = atomicAdd(&cursor[s], 1);
    atomList[offsets[s] + p] = a;
}

// ---------------------------------------------------------------------------
// WMMA GEMM over 16-atom tiles of one species.
//   mode 0: out = celu(A@W + bias); store f16 act, f32 celu' (and f32 act opt)
//   mode 1: out = (A@W) * derivIn;  store f16 (backward through celu)
//   mode 2: out = A@W;              store f32 (final d(aev))
// Block = 1 wave (32 threads) -> EXEC all ones at every WMMA.
// A: two aligned b128 loads per K-step (per ISA 7.12.2 16-bit A layout).
// B: fragment-packed buffer -> one aligned v16h (2x b128) load per K-step.
// Padding rows read a zeroed dummy row: no EXEC divergence in the hot loop.
// ---------------------------------------------------------------------------
__global__ __launch_bounds__(32)
void k_wmma_gemm(const _Float16* __restrict__ A, int lda, int K,
                 const _Float16* __restrict__ Wp, // fragment-packed
                 const float* __restrict__ bias, int biasLd,
                 const float* __restrict__ derivIn, int lddi,
                 _Float16* __restrict__ outH, float* __restrict__ outD,
                 float* __restrict__ outF, int ldo,
                 const _Float16* __restrict__ zeroRow,
                 const int* __restrict__ tileS, const int* __restrict__ tileBase,
                 const int* __restrict__ nTiles, const int* __restrict__ atomList,
                 const int* __restrict__ spOffsets, const int* __restrict__ spCounts,
                 int mode) {
    int t = blockIdx.x;
    if (t >= nTiles[0]) return;
    int s = tileS[t];
    int base = tileBase[t];
    int end = spOffsets[s] + spCounts[s];

    int lane = threadIdx.x;
    int sub = lane & 15;
    int half8 = (lane >> 4) * 8;

    int rowA = base + sub;
    int atomA = (rowA < end) ? atomList[rowA] : -1;
    const _Float16* Arow = (atomA >= 0) ? (A + (size_t)atomA * lda) : zeroRow;

    int nK0 = K >> 5;
    const _Float16* Bp = Wp + ((((size_t)s * gridDim.y + blockIdx.y) * nK0) * 32 + lane) * 16;

    v8f cacc;
#pragma unroll
    for (int r = 0; r < 8; ++r) cacc[r] = 0.0f;

    union AFrag { v16h v; v8h h[2]; };

#pragma unroll 2
    for (int k0 = 0; k0 < K; k0 += 32) {
        AFrag af;
        af.h[0] = *reinterpret_cast<const v8h*>(Arow + k0 + half8);
        af.h[1] = *reinterpret_cast<const v8h*>(Arow + k0 + 16 + half8);
        v16h b = *reinterpret_cast<const v16h*>(Bp);
        Bp += 32 * 16;
        cacc = __builtin_amdgcn_wmma_f32_16x16x32_f16(
            false, af.v, false, b, (short)0, cacc, false, false);
    }

    int col = blockIdx.y * 16 + sub;
    int half = lane >> 4;
#pragma unroll
    for (int r = 0; r < 8; ++r) {
        int m = r + 8 * half;
        int row = base + m;
        if (row >= end) continue;
        int atom = atomList[row];
        size_t o = (size_t)atom * ldo + col;
        float v = cacc[r];
        if (mode == 0) {
            v += bias[s * biasLd + col];
            float h, d;
            if (v > 0.0f) { h = v; d = 1.0f; }
            else          { float ex = expf(v * 10.0f); h = 0.1f * (ex - 1.0f); d = ex; }
            outH[o] = (_Float16)h;
            outD[o] = d;
            if (outF) outF[o] = h;
        } else if (mode == 1) {
            float g = v * derivIn[(size_t)atom * lddi + col];
            outH[o] = (_Float16)g;
        } else {
            outF[o] = v;
        }
    }
}

// ---------------------------------------------------------------------------
// layer 4 + energy reduction
// ---------------------------------------------------------------------------
__global__ void k_l4_energy(const float* __restrict__ h3F, const float* __restrict__ W4,
                            const float* __restrict__ b4, const float* __restrict__ sae,
                            const int* __restrict__ sgp, float* Eacc, int N) {
    int a = blockIdx.x * blockDim.x + threadIdx.x;
    if (a >= N) return;
    int g = sgp[a];
    bool valid = (g >= 0);
    int sp = clamp_sp(g);
    const float* h = h3F + (size_t)a * H3D;
    const float* w = W4 + sp * H3D;
    float e = 0.0f;
    for (int k = 0; k < H3D; ++k) e += h[k] * w[k];
    e += b4[sp] + sae[sp];
    atomicAdd(Eacc, valid ? e : 0.0f);
}

// backward seed: dE/dz3 = (valid ? W4 : 0) * celu'(z3)
__global__ void k_bwd_g3(const float* __restrict__ d3F, const float* __restrict__ W4,
                         const int* __restrict__ sgp, _Float16* __restrict__ g3H, int N) {
    size_t idx = (size_t)blockIdx.x * blockDim.x + threadIdx.x;
    if (idx >= (size_t)N * H3D) return;
    int a = (int)(idx / H3D);
    int k = (int)(idx % H3D);
    int g = sgp[a];
    float v = (g >= 0) ? W4[clamp_sp(g) * H3D + k] * d3F[idx] : 0.0f;
    g3H[idx] = (_Float16)v;
}

// ---------------------------------------------------------------------------
// radial backward: forces from d(aev)/radial section
// ---------------------------------------------------------------------------
__global__ void k_radial_bwd(const float* __restrict__ c, const int* __restrict__ sp,
                             const int* __restrict__ ai, const unsigned char* __restrict__ em,
                             const float* __restrict__ daev, float* __restrict__ grad, int E) {
    int e = blockIdx.x * blockDim.x + threadIdx.x;
    if (e >= E || !em[e]) return;
    int i = ai[e], j = ai[E + e];
    float dx = c[3 * j + 0] - c[3 * i + 0];
    float dy = c[3 * j + 1] - c[3 * i + 1];
    float dz = c[3 * j + 2] - c[3 * i + 2];
    float d = sqrtf(dx * dx + dy * dy + dz * dz + 1e-12f);
    if (d >= RCR_C) return;
    float fc = 0.5f * cosf(PI_F * d / RCR_C) + 0.5f;
    float dfc = -0.5f * (PI_F / RCR_C) * sinf(PI_F * d / RCR_C);
    int si = sp[i], sj = sp[j];
    const float* gi = daev + (size_t)i * AEV_PAD + sj * 16;
    const float* gj = daev + (size_t)j * AEV_PAD + si * 16;
    float S = 0.0f;
#pragma unroll
    for (int m = 0; m < 16; ++m) {
        float t = d - (0.8f + 0.26875f * m);
        float ex = 0.25f * expf(-ETAR_C * t * t);
        float drdd = ex * (-2.0f * ETAR_C * t * fc + dfc);
        S += (gi[m] + gj[m]) * drdd;
    }
    float inv = 1.0f / d;
    float gx = S * dx * inv, gy = S * dy * inv, gz = S * dz * inv;
    atomicAdd(&grad[3 * j + 0], gx);
    atomicAdd(&grad[3 * j + 1], gy);
    atomicAdd(&grad[3 * j + 2], gz);
    atomicAdd(&grad[3 * i + 0], -gx);
    atomicAdd(&grad[3 * i + 1], -gy);
    atomicAdd(&grad[3 * i + 2], -gz);
}

// ---------------------------------------------------------------------------
// angular backward: analytic gradient of the angular terms
// ---------------------------------------------------------------------------
__global__ void k_angular_bwd(const float* __restrict__ c, const int* __restrict__ sp,
                              const int* __restrict__ nbrj, const unsigned char* __restrict__ nm,
                              const float* __restrict__ daev, float* __restrict__ grad) {
    int n = blockIdx.x;
    int tid = threadIdx.x;
    __shared__ float vx[KNBR], vy[KNBR], vz[KNBR], dn[KNBR], fc[KNBR], fcd[KNBR];
    __shared__ int spj[KNBR], nid[KNBR];
    __shared__ float gx[KNBR], gy[KNBR], gz[KNBR];

    if (tid < KNBR) {
        int j = nbrj[n * KNBR + tid];
        float x = c[3 * j + 0] - c[3 * n + 0];
        float y = c[3 * j + 1] - c[3 * n + 1];
        float z = c[3 * j + 2] - c[3 * n + 2];
        float d = sqrtf(x * x + y * y + z * z + 1e-12f);
        bool in = (d < RCA_C) && nm[n * KNBR + tid];
        float f  = in ? (0.5f * cosf(PI_F * d / RCA_C) + 0.5f) : 0.0f;
        float fd = in ? (-0.5f * (PI_F / RCA_C) * sinf(PI_F * d / RCA_C)) : 0.0f;
        vx[tid] = x; vy[tid] = y; vz[tid] = z; dn[tid] = d;
        fc[tid] = f; fcd[tid] = fd;
        spj[tid] = sp[j]; nid[tid] = j;
        gx[tid] = 0.0f; gy[tid] = 0.0f; gz[tid] = 0.0f;
    }
    __syncthreads();

    const float* dv = daev + (size_t)n * AEV_PAD + RAD_OFF;

    for (int pid = tid; pid < NPAIR; pid += blockDim.x) {
        int rem = pid, jj = 0;
        while (rem >= (KNBR - 1) - jj) { rem -= (KNBR - 1) - jj; ++jj; }
        int kk = jj + 1 + rem;
        float fj = fc[jj], fk = fc[kk];
        float bse = 2.0f * fj * fk;
        if (bse == 0.0f) continue;
        float dj = dn[jj], dk = dn[kk];
        float dot = vx[jj] * vx[kk] + vy[jj] * vy[kk] + vz[jj] * vz[kk];
        float u = dot / fmaxf(dj * dk, 1e-10f);
        float q = 0.95f * u;
        float th = acosf(q);
        float davg = 0.5f * (dj + dk);
        int a = min(spj[jj], spj[kk]);
        int b = max(spj[jj], spj[kk]);
        int p = a * NSPC - (a * (a - 1)) / 2 + (b - a);
        const float* g = dv + p * 32;

        float f2[8], df2[8];
#pragma unroll
        for (int aa = 0; aa < 8; ++aa) {
            float t = davg - (0.8f + 0.3375f * aa);
            f2[aa] = expf(-ETAA_C * t * t);
            df2[aa] = -2.0f * ETAA_C * t * f2[aa];
        }
        float S1 = 0.0f, Sth = 0.0f, Sd = 0.0f;
#pragma unroll
        for (int z = 0; z < 4; ++z) {
            float ang = th - (z + 0.5f) * (PI_F / 4.0f);
            float ch = cosf(ang), sh = sinf(ang);
            float hz = fmaxf(0.5f * (1.0f + ch), 0.0f);
            float f1 = powf(hz, ZETA_C);
            float df1 = ZETA_C * powf(hz, ZETA_C - 1.0f) * (-0.5f * sh);
#pragma unroll
            for (int aa = 0; aa < 8; ++aa) {
                float gv = g[z * 8 + aa];
                S1  += gv * f1 * f2[aa];
                Sth += gv * df1 * f2[aa];
                Sd  += gv * f1 * df2[aa];
            }
        }
        float dthdu = -0.95f / sqrtf(fmaxf(1.0f - q * q, 1e-6f));
        float invjk = 1.0f / fmaxf(dj * dk, 1e-10f);
        float cth = bse * Sth * dthdu;

        float Aj = S1 * 2.0f * fk * fcd[jj] / dj + bse * Sd * 0.5f / dj - cth * u / (dj * dj);
        float Ak = S1 * 2.0f * fj * fcd[kk] / dk + bse * Sd * 0.5f / dk - cth * u / (dk * dk);
        float B  = cth * invjk;

        float gjx = Aj * vx[jj] + B * vx[kk];
        float gjy = Aj * vy[jj] + B * vy[kk];
        float gjz = Aj * vz[jj] + B * vz[kk];
        float gkx = Ak * vx[kk] + B * vx[jj];
        float gky = Ak * vy[kk] + B * vy[jj];
        float gkz = Ak * vz[kk] + B * vz[jj];

        atomicAdd(&gx[jj], gjx); atomicAdd(&gy[jj], gjy); atomicAdd(&gz[jj], gjz);
        atomicAdd(&gx[kk], gkx); atomicAdd(&gy[kk], gky); atomicAdd(&gz[kk], gkz);
    }
    __syncthreads();

    if (tid < KNBR) {
        atomicAdd(&grad[3 * nid[tid] + 0], gx[tid]);
        atomicAdd(&grad[3 * nid[tid] + 1], gy[tid]);
        atomicAdd(&grad[3 * nid[tid] + 2], gz[tid]);
    }
    if (tid == 0) {
        float sx = 0.0f, sy = 0.0f, sz = 0.0f;
        for (int i = 0; i < KNBR; ++i) { sx += gx[i]; sy += gy[i]; sz += gz[i]; }
        atomicAdd(&grad[3 * n + 0], -sx);
        atomicAdd(&grad[3 * n + 1], -sy);
        atomicAdd(&grad[3 * n + 2], -sz);
    }
}

// ---------------------------------------------------------------------------
// final packing: out[0] = E, out[1..] = -grad
// ---------------------------------------------------------------------------
__global__ void k_finalize(const float* __restrict__ Eacc, const float* __restrict__ grad,
                           float* __restrict__ out, int total) {
    int i = blockIdx.x * blockDim.x + threadIdx.x;
    if (i >= total) return;
    out[i] = (i == 0) ? Eacc[0] : -grad[i - 1];
}

// ---------------------------------------------------------------------------
// host launcher
// ---------------------------------------------------------------------------
extern "C" void kernel_launch(void* const* d_in, const int* in_sizes, int n_in,
                              void* d_out, int out_size, void* d_ws, size_t ws_size,
                              hipStream_t stream) {
    const int*   species = (const int*)d_in[0];
    const float* coords  = (const float*)d_in[1];
    const int*   ai12    = (const int*)d_in[2];
    const int*   sgp     = (const int*)d_in[5];
    const unsigned char* emask = (const unsigned char*)d_in[6];
    const int*   nbrj    = (const int*)d_in[7];
    const unsigned char* nmask = (const unsigned char*)d_in[8];
    const float* W1 = (const float*)d_in[9];
    const float* b1 = (const float*)d_in[10];
    const float* W2 = (const float*)d_in[11];
    const float* b2 = (const float*)d_in[12];
    const float* W3 = (const float*)d_in[13];
    const float* b3 = (const float*)d_in[14];
    const float* W4 = (const float*)d_in[15];
    const float* b4 = (const float*)d_in[16];
    const float* sae = (const float*)d_in[17];

    const int N = in_sizes[0];   // 2048 atoms
    const int E = in_sizes[6];   // 65536 edges

    // ---- workspace carve-out ----
    char* wsp = (char*)d_ws;
    size_t off = 0;
    auto alloc = [&](size_t bytes) -> char* {
        char* p = wsp + off;
        off += (bytes + 255) & ~size_t(255);
        return p;
    };
    float*    aevF  = (float*)alloc((size_t)N * AEV_PAD * 4);
    _Float16* aevH  = (_Float16*)alloc((size_t)N * AEV_PAD * 2);
    _Float16* h1H   = (_Float16*)alloc((size_t)N * H1D * 2);
    float*    d1F   = (float*)alloc((size_t)N * H1D * 4);
    _Float16* h2H   = (_Float16*)alloc((size_t)N * H2D * 2);
    float*    d2F   = (float*)alloc((size_t)N * H2D * 4);
    _Float16* h3H   = (_Float16*)alloc((size_t)N * H3D * 2);
    float*    h3F   = (float*)alloc((size_t)N * H3D * 4);
    float*    d3F   = (float*)alloc((size_t)N * H3D * 4);
    _Float16* g3H   = (_Float16*)alloc((size_t)N * H3D * 2);
    _Float16* g2H   = (_Float16*)alloc((size_t)N * H2D * 2);
    _Float16* g1H   = (_Float16*)alloc((size_t)N * H1D * 2);
    float*    daevF = (float*)alloc((size_t)N * AEV_PAD * 4);
    float*    grad  = (float*)alloc((size_t)N * 3 * 4);
    float*    Eacc  = (float*)alloc(4);
    _Float16* zeroRow = (_Float16*)alloc((size_t)AEV_PAD * 2);
    int* counts   = (int*)alloc(NSPC * 4);
    int* offsets  = (int*)alloc(NSPC * 4);
    int* cursor   = (int*)alloc(NSPC * 4);
    int* tileS    = (int*)alloc(MAX_TILES * 4);
    int* tileBase = (int*)alloc(MAX_TILES * 4);
    int* nTiles   = (int*)alloc(4);
    int* atomList = (int*)alloc((size_t)N * 4);
    _Float16* W1P  = (_Float16*)alloc((size_t)NSPC * AEV_PAD * H1D * 2);
    _Float16* W2P  = (_Float16*)alloc((size_t)NSPC * H1D * H2D * 2);
    _Float16* W3P  = (_Float16*)alloc((size_t)NSPC * H2D * H3D * 2);
    _Float16* W1TP = (_Float16*)alloc((size_t)NSPC * H1D * 1008 * 2);
    _Float16* W2TP = (_Float16*)alloc((size_t)NSPC * H2D * H1D * 2);
    _Float16* W3TP = (_Float16*)alloc((size_t)NSPC * H3D * H2D * 2);
    (void)ws_size; (void)n_in;

    // ---- zero accumulators (graph-capture-safe async memsets) ----
    (void)hipMemsetAsync(aevF, 0, (size_t)N * AEV_PAD * 4, stream);
    (void)hipMemsetAsync(grad, 0, (size_t)N * 3 * 4, stream);
    (void)hipMemsetAsync(Eacc, 0, 4, stream);
    (void)hipMemsetAsync(zeroRow, 0, (size_t)AEV_PAD * 2, stream);
    (void)hipMemsetAsync(counts, 0, NSPC * 4, stream);
    (void)hipMemsetAsync(cursor, 0, NSPC * 4, stream);

    const int TB = 256;
    auto blocks = [](size_t total, int tb) { return (unsigned)((total + tb - 1) / tb); };

    // ---- pack weights into fp16 WMMA fragment order ----
    // forward: B = W (K x N), K padded to mult of 32
    k_packB<<<blocks((size_t)NSPC * AEV_PAD * H1D, TB), TB, 0, stream>>>(
        W1, W1P, NSPC, 1008, H1D, 0, 1008, AEV_PAD / 32, H1D / 16);
    k_packB<<<blocks((size_t)NSPC * H1D * H2D, TB), TB, 0, stream>>>(
        W2, W2P, NSPC, H1D, H2D, 0, H1D, H1D / 32, H2D / 16);
    k_packB<<<blocks((size_t)NSPC * H2D * H3D, TB), TB, 0, stream>>>(
        W3, W3P, NSPC, H2D, H3D, 0, H2D, H2D / 32, H3D / 16);
    // backward: B = W^T (K x N)
    k_packB<<<blocks((size_t)NSPC * H1D * 1008, TB), TB, 0, stream>>>(
        W1, W1TP, NSPC, 1008, H1D, 1, H1D, H1D / 32, 1008 / 16);
    k_packB<<<blocks((size_t)NSPC * H2D * H1D, TB), TB, 0, stream>>>(
        W2, W2TP, NSPC, H1D, H2D, 1, H2D, H2D / 32, H1D / 16);
    k_packB<<<blocks((size_t)NSPC * H3D * H2D, TB), TB, 0, stream>>>(
        W3, W3TP, NSPC, H2D, H3D, 1, H3D, H3D / 32, H2D / 16);

    // ---- AEV forward ----
    k_radial_fwd<<<blocks(E, TB), TB, 0, stream>>>(coords, species, ai12, emask, aevF, E);
    k_angular_fwd<<<N, 128, 0, stream>>>(coords, species, nbrj, nmask, aevF);
    k_f2h<<<blocks((size_t)N * AEV_PAD, TB), TB, 0, stream>>>(aevF, aevH, (size_t)N * AEV_PAD);

    // ---- species bucketing + tile table ----
    k_count<<<blocks(N, TB), TB, 0, stream>>>(species, counts, N);
    k_build_tiles<<<1, 1, 0, stream>>>(counts, offsets, tileS, tileBase, nTiles);
    k_fill<<<blocks(N, TB), TB, 0, stream>>>(species, offsets, cursor, atomList, N);

    // ---- MLP forward (WMMA) ----
    k_wmma_gemm<<<dim3(MAX_TILES, H1D / 16), 32, 0, stream>>>(
        aevH, AEV_PAD, AEV_PAD, W1P, b1, H1D, nullptr, 0,
        h1H, d1F, nullptr, H1D, zeroRow, tileS, tileBase, nTiles, atomList, offsets, counts, 0);
    k_wmma_gemm<<<dim3(MAX_TILES, H2D / 16), 32, 0, stream>>>(
        h1H, H1D, H1D, W2P, b2, H2D, nullptr, 0,
        h2H, d2F, nullptr, H2D, zeroRow, tileS, tileBase, nTiles, atomList, offsets, counts, 0);
    k_wmma_gemm<<<dim3(MAX_TILES, H3D / 16), 32, 0, stream>>>(
        h2H, H2D, H2D, W3P, b3, H3D, nullptr, 0,
        h3H, d3F, h3F, H3D, zeroRow, tileS, tileBase, nTiles, atomList, offsets, counts, 0);

    // ---- layer 4 + energy ----
    k_l4_energy<<<blocks(N, TB), TB, 0, stream>>>(h3F, W4, b4, sae, sgp, Eacc, N);

    // ---- MLP backward (WMMA against transposed weights) ----
    k_bwd_g3<<<blocks((size_t)N * H3D, TB), TB, 0, stream>>>(d3F, W4, sgp, g3H, N);
    k_wmma_gemm<<<dim3(MAX_TILES, H2D / 16), 32, 0, stream>>>(
        g3H, H3D, H3D, W3TP, nullptr, 0, d2F, H2D,
        g2H, nullptr, nullptr, H2D, zeroRow, tileS, tileBase, nTiles, atomList, offsets, counts, 1);
    k_wmma_gemm<<<dim3(MAX_TILES, H1D / 16), 32, 0, stream>>>(
        g2H, H2D, H2D, W2TP, nullptr, 0, d1F, H1D,
        g1H, nullptr, nullptr, H1D, zeroRow, tileS, tileBase, nTiles, atomList, offsets, counts, 1);
    k_wmma_gemm<<<dim3(MAX_TILES, 1008 / 16), 32, 0, stream>>>(
        g1H, H1D, H1D, W1TP, nullptr, 0, nullptr, 0,
        nullptr, nullptr, daevF, AEV_PAD, zeroRow, tileS, tileBase, nTiles, atomList, offsets, counts, 2);

    // ---- AEV backward -> coordinate gradient ----
    k_radial_bwd<<<blocks(E, TB), TB, 0, stream>>>(coords, species, ai12, emask, daevF, grad, E);
    k_angular_bwd<<<N, 128, 0, stream>>>(coords, species, nbrj, nmask, daevF, grad);

    // ---- pack outputs: E then forces = -grad ----
    k_finalize<<<blocks(out_size, TB), TB, 0, stream>>>(Eacc, grad, (float*)d_out, out_size);
}
